// LlamaAttentionWithLora_28286654611999
// MI455X (gfx1250) — matrix operand order repaired and checked by reading, not compiled, not run on hardware
//
#include <hip/hip_runtime.h>

#define H   32
#define D   128
#define HID 4096
#define DOFF 1024
#define BD  32
#define T   1056
#define MAXKV 513
#define R   16
#define NA  4
#define SCALE 0.08838834764831845f

typedef __attribute__((ext_vector_type(16))) __bf16 v16bf;
typedef __attribute__((ext_vector_type(8)))  float  v8f;

union Frag { v16bf v; unsigned int u[8]; };

// RNE-biased value kept in a full 32-bit register (bf16 = high half)
__device__ __forceinline__ unsigned int bf16_rne_bits(float f) {
    unsigned int u = __float_as_uint(f);
    return u + 0x7FFFu + ((u >> 16) & 1u);
}
// pack two bf16 (from biased f32 bit patterns) with one v_perm_b32
__device__ __forceinline__ unsigned int pack_bf16(float lo, float hi) {
    return __builtin_amdgcn_perm(bf16_rne_bits(hi), bf16_rne_bits(lo), 0x07060302u);
}
__device__ __forceinline__ unsigned short f32_to_bf16(float f) {
    return (unsigned short)(bf16_rne_bits(f) >> 16);
}

// ---- CDNA5 async memory->LDS copy (ASYNCcnt path, 16B per lane) -------------
__device__ __forceinline__ void async_copy_b128(const void* gptr, void* lds_ptr) {
    unsigned lds_lo = (unsigned)(size_t)lds_ptr;       // low 32b of generic = LDS addr
    unsigned long long ga = (unsigned long long)(size_t)gptr;
    asm volatile("global_load_async_to_lds_b128 %0, %1, off"
                 :: "v"(lds_lo), "v"(ga) : "memory");
}
__device__ __forceinline__ void wait_async0() {
    asm volatile("s_wait_asynccnt 0" ::: "memory");
}

// ---------------------------------------------------------------- f32 -> bf16
__global__ __launch_bounds__(256) void cvt_bf16_kernel(const float* __restrict__ in,
                                                       unsigned int* __restrict__ out,
                                                       int npair) {
    int i = blockIdx.x * 256 + threadIdx.x;
    if (i < npair) {
        float2 v = ((const float2*)in)[i];
        out[i] = pack_bf16(v.x, v.y);
    }
}

// -------------------------------------------- C[M,N] = A_bf16[M,K] @ B_f32[K,N]
// 128x128 tile, 8 waves, each wave 64x32 (4x2 wmma 16x16x32 bf16 tiles).
// Double-buffered LDS; A staged with global_load_async_to_lds_b128.
__global__ __launch_bounds__(256) void gemm_bf16_kernel(const unsigned short* __restrict__ A,
                                                        const float* __restrict__ B,
                                                        float* __restrict__ C,
                                                        int M, int N, int K) {
    __shared__ unsigned short As[2][128 * 32];   // [m][k]
    __shared__ unsigned short Bs[2][128 * 32];   // [n][k]  (k pairs contiguous)

    const int tid  = threadIdx.x;
    const int lane = tid & 31;
    const int w    = tid >> 5;
    const int m0   = blockIdx.x * 128;
    const int n0   = blockIdx.y * 128;
    const int mw   = w & 1, nw = w >> 1;
    const int lm   = lane & 15, lh = lane >> 4;
    const int nt   = K / 32;

    // A staging: thread covers one 32B row-chunk -> two async b128 copies
    auto stageA = [&](int kt, int buf) {
        int r = tid >> 1, halfi = tid & 1;
        int grow = m0 + r;
        if (grow >= M) grow = M - 1;     // duplicate last row; feeds only unstored C rows
        const unsigned short* gp = A + (size_t)grow * K + kt * 32 + halfi * 16;
        unsigned short* lp = &As[buf][r * 32 + halfi * 16];
        async_copy_b128(gp, lp);
        async_copy_b128(gp + 8, lp + 8);
    };

    // B staging: thread owns a 2(k) x 8(n) patch; f32 rows held in regs,
    // converted + packed to bf16 pairs after the compute phase.
    float br0[8], br1[8];
    auto loadB = [&](int kt) {
        int k2 = (tid >> 4) << 1;
        int ng = (tid & 15) << 3;
        const float* src = B + (size_t)(kt * 32 + k2) * N + n0 + ng;
        const float4* s0 = (const float4*)src;
        const float4* s1 = (const float4*)(src + N);
        float4 a0 = s0[0], a1 = s0[1], b0 = s1[0], b1 = s1[1];
        br0[0]=a0.x; br0[1]=a0.y; br0[2]=a0.z; br0[3]=a0.w;
        br0[4]=a1.x; br0[5]=a1.y; br0[6]=a1.z; br0[7]=a1.w;
        br1[0]=b0.x; br1[1]=b0.y; br1[2]=b0.z; br1[3]=b0.w;
        br1[4]=b1.x; br1[5]=b1.y; br1[6]=b1.z; br1[7]=b1.w;
    };
    auto storeB = [&](int buf) {
        int k2 = (tid >> 4) << 1;
        int ng = (tid & 15) << 3;
#pragma unroll
        for (int j = 0; j < 8; ++j)
            *(unsigned int*)&Bs[buf][(ng + j) * 32 + k2] = pack_bf16(br0[j], br1[j]);
    };

    v8f zero = {0.f,0.f,0.f,0.f,0.f,0.f,0.f,0.f};
    v8f acc[4][2];
#pragma unroll
    for (int i = 0; i < 4; ++i)
#pragma unroll
        for (int j = 0; j < 2; ++j) acc[i][j] = zero;

    // prologue
    stageA(0, 0);
    loadB(0);
    storeB(0);
    wait_async0();
    __syncthreads();

    int cb = 0;
    for (int kt = 0; kt < nt; ++kt) {
        int nb = cb ^ 1;
        bool more = (kt + 1 < nt);
        if (more) { stageA(kt + 1, nb); loadB(kt + 1); }

        Frag a[4], b[2];
#pragma unroll
        for (int sm = 0; sm < 4; ++sm) {
            int m = mw * 64 + sm * 16 + lm;
#pragma unroll
            for (int i = 0; i < 8; ++i) {
                int k = ((i >> 2) << 4) + (lh << 3) + ((i & 3) << 1);
                a[sm].u[i] = *(const unsigned int*)&As[cb][m * 32 + k];
            }
        }
#pragma unroll
        for (int sn = 0; sn < 2; ++sn) {
            int n = nw * 32 + sn * 16 + lm;
#pragma unroll
            for (int i = 0; i < 8; ++i) {
                int k = ((i >> 2) << 4) + (lh << 3) + ((i & 3) << 1);
                b[sn].u[i] = *(const unsigned int*)&Bs[cb][n * 32 + k];
            }
        }
#pragma unroll
        for (int sm = 0; sm < 4; ++sm)
#pragma unroll
            for (int sn = 0; sn < 2; ++sn)
                acc[sm][sn] = __builtin_amdgcn_wmma_f32_16x16x32_bf16(
                    false, a[sm].v, false, b[sn].v, (short)0, acc[sm][sn], false, false);

        if (more) { storeB(nb); wait_async0(); }
        __syncthreads();
        cb = nb;
    }

#pragma unroll
    for (int sm = 0; sm < 4; ++sm)
#pragma unroll
        for (int sn = 0; sn < 2; ++sn) {
            int n = n0 + nw * 32 + sn * 16 + lm;
#pragma unroll
            for (int j = 0; j < 8; ++j) {
                int m = m0 + mw * 64 + sm * 16 + (lh << 3) + j;
                if (m < M) C[(size_t)m * N + n] = acc[sm][sn][j];
            }
        }
}

// ------------------------------------------------- segmented LoRA: y += x@A@B
__global__ __launch_bounds__(256) void lora_kernel(const float* __restrict__ x,
                                                   const float* __restrict__ wa,
                                                   const float* __restrict__ wb,
                                                   const int* __restrict__ segment,
                                                   float* __restrict__ y) {
    __shared__ float red[256 * R];
    __shared__ float u[R];
    const int t = blockIdx.x, tid = threadIdx.x;

    int aid = 0;
#pragma unroll
    for (int i = 1; i < 5; ++i) if (segment[i] <= t) aid = i;
    if (aid > NA - 1) aid = NA - 1;

    float acc[R];
#pragma unroll
    for (int r = 0; r < R; ++r) acc[r] = 0.f;
    for (int hh = tid; hh < HID; hh += 256) {
        float xv = x[(size_t)t * HID + hh];
        const float* wp = wa + ((size_t)aid * HID + hh) * R;
#pragma unroll
        for (int r = 0; r < R; ++r) acc[r] += xv * wp[r];
    }
#pragma unroll
    for (int r = 0; r < R; ++r) red[tid * R + r] = acc[r];
    __syncthreads();
    for (int s = 128; s > 0; s >>= 1) {
        if (tid < s)
#pragma unroll
            for (int r = 0; r < R; ++r) red[tid * R + r] += red[(tid + s) * R + r];
        __syncthreads();
    }
    if (tid < R) u[tid] = red[tid];
    __syncthreads();

    for (int c = tid; c < HID; c += 256) {
        float s = 0.f;
        const float* wp = wb + (size_t)aid * R * HID + c;
#pragma unroll
        for (int r = 0; r < R; ++r) s += u[r] * wp[(size_t)r * HID];
        y[(size_t)t * HID + c] += s;
    }
}

// ------------------------- prefill attention, WMMA for QK^T and P*V, S in LDS
// grid (H, DOFF/32), block 256 (8 waves), dynamic LDS.
#define QB 32
__global__ __launch_bounds__(256) void prefill_attn_kernel(const float* __restrict__ qp,
                                                           const float* __restrict__ kp,
                                                           const float* __restrict__ vp,
                                                           const int* __restrict__ indptr,
                                                           float* __restrict__ attn) {
    extern __shared__ char smem[];
    float*          S   = (float*)smem;                         // [QB][DOFF]
    unsigned short* Qs  = (unsigned short*)(S + QB * DOFF);     // [QB][128]
    unsigned short* Ks  = Qs + QB * 128;                        // [64][128]
    unsigned short* Vs  = Ks + 64 * 128;                        // [128][64] (transposed)
    float*          red = (float*)(Vs + 128 * 64);              // [QB][8]

    const int h = blockIdx.x, q0 = blockIdx.y * QB;
    const int tid = threadIdx.x, lane = tid & 31, w = tid >> 5;
    const int lm = lane & 15, lh = lane >> 4;
    const int ip1 = indptr[1], ip2 = indptr[2], ip3 = indptr[3];

    // stage Q block as packed bf16 pairs
    for (int idx = tid; idx < QB * 64; idx += 256) {
        int r = idx >> 6, dp = (idx & 63) << 1;
        float2 v = *(const float2*)&qp[(size_t)(q0 + r) * HID + h * D + dp];
        *(unsigned int*)&Qs[r * 128 + dp] = pack_bf16(v.x, v.y);
    }

    // ---- phase 1: S = Q K^T (raw, unscaled) ----
    const int sm = w & 1, sn4 = w >> 1;
    for (int kb = 0; kb < DOFF; kb += 64) {
        for (int idx = tid; idx < 64 * 64; idx += 256) {
            int kk = idx >> 6, dp = (idx & 63) << 1;
            float2 v = *(const float2*)&kp[(size_t)(kb + kk) * HID + h * D + dp];
            *(unsigned int*)&Ks[kk * 128 + dp] = pack_bf16(v.x, v.y);
        }
        __syncthreads();

        v8f acc = {0.f,0.f,0.f,0.f,0.f,0.f,0.f,0.f};
#pragma unroll
        for (int dk = 0; dk < 128; dk += 32) {
            Frag a, b;
            int m = sm * 16 + lm;
#pragma unroll
            for (int i = 0; i < 8; ++i) {
                int d = dk + ((i >> 2) << 4) + (lh << 3) + ((i & 3) << 1);
                a.u[i] = *(const unsigned int*)&Qs[m * 128 + d];
            }
            int kkey = sn4 * 16 + lm;
#pragma unroll
            for (int i = 0; i < 8; ++i) {
                int d = dk + ((i >> 2) << 4) + (lh << 3) + ((i & 3) << 1);
                b.u[i] = *(const unsigned int*)&Ks[kkey * 128 + d];
            }
            acc = __builtin_amdgcn_wmma_f32_16x16x32_bf16(false, a.v, false, b.v,
                                                          (short)0, acc, false, false);
        }
        int n = sn4 * 16 + lm;
#pragma unroll
        for (int j = 0; j < 8; ++j) {
            int m = sm * 16 + (lh << 3) + j;
            S[m * DOFF + kb + n] = acc[j];
        }
        __syncthreads();
    }

    // ---- phase 2: masked softmax over rows of S ----
    {
        const int row = tid >> 3, sl = tid & 7;
        const int qg = q0 + row;
        int sq = 0; if (ip1 <= qg) sq = 1; if (ip2 <= qg) sq = 2; if (ip3 <= qg) sq = 3;

        float mx = -3e38f;
        for (int c = sl; c < DOFF; c += 8) {
            int sk = 0; if (ip1 <= c) sk = 1; if (ip2 <= c) sk = 2; if (ip3 <= c) sk = 3;
            float val = (sk == sq && c <= qg) ? S[row * DOFF + c] * SCALE : -3e38f;
            S[row * DOFF + c] = val;
            mx = fmaxf(mx, val);
        }
        red[row * 8 + sl] = mx;
        __syncthreads();
        float rmax = red[row * 8];
#pragma unroll
        for (int j = 1; j < 8; ++j) rmax = fmaxf(rmax, red[row * 8 + j]);
        __syncthreads();

        float sum = 0.f;
        for (int c = sl; c < DOFF; c += 8) {
            float p = __expf(S[row * DOFF + c] - rmax);
            S[row * DOFF + c] = p;
            sum += p;
        }
        red[row * 8 + sl] = sum;
        __syncthreads();
        float rsum = 0.f;
#pragma unroll
        for (int j = 0; j < 8; ++j) rsum += red[row * 8 + j];
        float inv = 1.f / rsum;
        for (int c = sl; c < DOFF; c += 8) S[row * DOFF + c] *= inv;
        __syncthreads();
    }

    // ---- phase 3: O = P V ----
    const int ndb = (w >> 1) << 1;
    v8f oacc[2];
    oacc[0] = (v8f){0.f,0.f,0.f,0.f,0.f,0.f,0.f,0.f};
    oacc[1] = oacc[0];
    for (int kb = 0; kb < DOFF; kb += 64) {
        __syncthreads();
        // stage V transposed [d][k], packed key-pairs
        for (int idx = tid; idx < 32 * 128; idx += 256) {
            int p = idx >> 7, d = idx & 127;
            int kk = p << 1;
            float f0 = vp[(size_t)(kb + kk) * HID + h * D + d];
            float f1 = vp[(size_t)(kb + kk + 1) * HID + h * D + d];
            *(unsigned int*)&Vs[d * 64 + kk] = pack_bf16(f0, f1);
        }
        __syncthreads();
#pragma unroll
        for (int dk = 0; dk < 64; dk += 32) {
            Frag a;
            int m = sm * 16 + lm;
#pragma unroll
            for (int i = 0; i < 8; ++i) {
                int kk = dk + ((i >> 2) << 4) + (lh << 3) + ((i & 3) << 1);
                a.u[i] = pack_bf16(S[m * DOFF + kb + kk], S[m * DOFF + kb + kk + 1]);
            }
#pragma unroll
            for (int nd = 0; nd < 2; ++nd) {
                Frag b;
                int d = (ndb + nd) * 16 + lm;
#pragma unroll
                for (int i = 0; i < 8; ++i) {
                    int kk = dk + ((i >> 2) << 4) + (lh << 3) + ((i & 3) << 1);
                    b.u[i] = *(const unsigned int*)&Vs[d * 64 + kk];
                }
                oacc[nd] = __builtin_amdgcn_wmma_f32_16x16x32_bf16(
                    false, a.v, false, b.v, (short)0, oacc[nd], false, false);
            }
        }
    }
#pragma unroll
    for (int nd = 0; nd < 2; ++nd) {
        int d = (ndb + nd) * 16 + lm;
#pragma unroll
        for (int j = 0; j < 8; ++j) {
            int m = sm * 16 + (lh << 3) + j;
            attn[(size_t)(q0 + m) * HID + h * D + d] = oacc[nd][j];
        }
    }
}

// ------------------------------------------- paged decode attention, one (b,h)
__global__ __launch_bounds__(128) void decode_attn_kernel(const float* __restrict__ qp,
                                                          const float* __restrict__ kp,
                                                          const float* __restrict__ vp,
                                                          const float* __restrict__ kc,
                                                          const float* __restrict__ vc,
                                                          const int* __restrict__ kv_lens,
                                                          float* __restrict__ attn) {
    __shared__ float qsh[D];
    __shared__ float sc[MAXKV + 7];
    __shared__ float rbuf[128];
    const int b = blockIdx.x, h = blockIdx.y, tid = threadIdx.x;
    const int t = DOFF + b;
    const int len = kv_lens[b] + 1;

    qsh[tid] = qp[(size_t)t * HID + h * D + tid];
    __syncthreads();

    for (int k = tid; k < len; k += 128) {
        const float* kv = (k == len - 1)
            ? (kp + (size_t)t * HID + h * D)
            : (kc + ((size_t)b * MAXKV + k) * (H * D) + h * D);
        float s = 0.f;
#pragma unroll 8
        for (int d = 0; d < D; ++d) s += qsh[d] * kv[d];
        sc[k] = s * SCALE;
    }
    __syncthreads();

    float mx = -3e38f;
    for (int k = tid; k < len; k += 128) mx = fmaxf(mx, sc[k]);
    rbuf[tid] = mx;
    __syncthreads();
    for (int s = 64; s > 0; s >>= 1) {
        if (tid < s) rbuf[tid] = fmaxf(rbuf[tid], rbuf[tid + s]);
        __syncthreads();
    }
    float rmax = rbuf[0];
    __syncthreads();

    float sum = 0.f;
    for (int k = tid; k < len; k += 128) {
        float p = __expf(sc[k] - rmax);
        sc[k] = p;
        sum += p;
    }
    rbuf[tid] = sum;
    __syncthreads();
    for (int s = 64; s > 0; s >>= 1) {
        if (tid < s) rbuf[tid] += rbuf[tid + s];
        __syncthreads();
    }
    float inv = 1.f / rbuf[0];
    __syncthreads();

    float acc = 0.f;
    for (int k = 0; k < len - 1; ++k)
        acc += sc[k] * vc[((size_t)b * MAXKV + k) * (H * D) + h * D + tid];
    acc += sc[len - 1] * vp[(size_t)t * HID + h * D + tid];
    attn[(size_t)t * HID + h * D + tid] = acc * inv;
}

// ----------------------------------------------------------------------------
extern "C" void kernel_launch(void* const* d_in, const int* in_sizes, int n_in,
                              void* d_out, int out_size, void* d_ws, size_t ws_size,
                              hipStream_t stream) {
    const float* hs   = (const float*)d_in[0];
    const float* Wq   = (const float*)d_in[1];
    const float* Wk   = (const float*)d_in[2];
    const float* Wv   = (const float*)d_in[3];
    const float* Wo   = (const float*)d_in[4];
    const float* wa_q = (const float*)d_in[5];
    const float* wb_q = (const float*)d_in[6];
    const float* wa_k = (const float*)d_in[7];
    const float* wb_k = (const float*)d_in[8];
    const float* wa_v = (const float*)d_in[9];
    const float* wb_v = (const float*)d_in[10];
    const float* wa_o = (const float*)d_in[11];
    const float* wb_o = (const float*)d_in[12];
    const float* kc   = (const float*)d_in[13];
    const float* vc   = (const float*)d_in[14];
    const int*   indptr  = (const int*)d_in[15];
    const int*   segment = (const int*)d_in[16];
    const int*   kv_lens = (const int*)d_in[17];
    float* out = (float*)d_out;

    const size_t nTH = (size_t)T * HID;
    char* wsb = (char*)d_ws;
    unsigned short* hsb   = (unsigned short*)wsb;  wsb += nTH * 2;
    float*          qp    = (float*)wsb;           wsb += nTH * 4;
    float*          kp    = (float*)wsb;           wsb += nTH * 4;
    float*          vp    = (float*)wsb;           wsb += nTH * 4;
    float*          attn  = (float*)wsb;           wsb += nTH * 4;
    unsigned short* attnb = (unsigned short*)wsb;

    const int smem_prefill = QB * DOFF * 4 + QB * 128 * 2 + 64 * 128 * 2 +
                             128 * 64 * 2 + QB * 8 * 4;
    (void)hipFuncSetAttribute((const void*)prefill_attn_kernel,
                              hipFuncAttributeMaxDynamicSharedMemorySize, smem_prefill);

    dim3 gGemm(9, HID / 128);
    const int npair = (int)(nTH / 2);

    cvt_bf16_kernel<<<(npair + 255) / 256, 256, 0, stream>>>(hs, (unsigned int*)hsb, npair);

    gemm_bf16_kernel<<<gGemm, 256, 0, stream>>>(hsb, Wq, qp, T, HID, HID);
    gemm_bf16_kernel<<<gGemm, 256, 0, stream>>>(hsb, Wk, kp, T, HID, HID);
    gemm_bf16_kernel<<<gGemm, 256, 0, stream>>>(hsb, Wv, vp, T, HID, HID);

    lora_kernel<<<T, 256, 0, stream>>>(hs, wa_q, wb_q, segment, qp);
    lora_kernel<<<T, 256, 0, stream>>>(hs, wa_k, wb_k, segment, kp);
    lora_kernel<<<T, 256, 0, stream>>>(hs, wa_v, wb_v, segment, vp);

    prefill_attn_kernel<<<dim3(H, DOFF / QB), 256, smem_prefill, stream>>>(
        qp, kp, vp, indptr, attn);
    decode_attn_kernel<<<dim3(BD, H), 128, 0, stream>>>(
        qp, kp, vp, kc, vc, kv_lens, attn);

    cvt_bf16_kernel<<<(npair + 255) / 256, 256, 0, stream>>>(attn, (unsigned int*)attnb, npair);
    gemm_bf16_kernel<<<gGemm, 256, 0, stream>>>(attnb, Wo, out, T, HID, HID);
    lora_kernel<<<T, 256, 0, stream>>>(attn, wa_o, wb_o, segment, out);
}
